// TemporalAttention_Pos_11897059410218
// MI455X (gfx1250) — matrix operand order repaired and checked by reading, not compiled, not run on hardware
//
#include <hip/hip_runtime.h>
#include <stdint.h>

// ---------------------------------------------------------------------------
// Types for CDNA5 WMMA (wave32): v_wmma_f32_16x16x32_bf16
// ---------------------------------------------------------------------------
typedef __attribute__((ext_vector_type(16))) __bf16 v16bf;
typedef __attribute__((ext_vector_type(8)))  float  v8f;

union Frag { uint4 u[2]; v16bf v; };

__device__ __forceinline__ unsigned short f2bf(float f) {
  // round-to-nearest-even fp32 -> bf16
  unsigned u = __float_as_uint(f);
  unsigned r = u + 0x7FFFu + ((u >> 16) & 1u);
  return (unsigned short)(r >> 16);
}

// A-operand (16-bit A 16x32, ISA 7.12.2): lane L: M = L%16, g = L/16.
// VGPR j<4 holds K = g*8 + 2j + {0,1}; j>=4 holds K = 16 + g*8 + 2(j-4) + {0,1}
// => two 16B chunks 32 bytes apart in a row-major [*,K] LDS row.
__device__ __forceinline__ v16bf ldsA(const unsigned short* p) {
  Frag f;
  f.u[0] = *(const uint4*)(p);
  f.u[1] = *(const uint4*)(p + 16);
  return f.v;
}
// B-operand (16-bit B 32x16): lane L: N = L%16, g = L/16, e<8 holds K = g*8 + e,
// e>=8 holds K = 16 + g*8 + (e-8) => two 16B chunks 32 bytes apart.
__device__ __forceinline__ v16bf ldsB(const unsigned short* p) {
  Frag f;
  f.u[0] = *(const uint4*)(p);
  f.u[1] = *(const uint4*)(p + 16);
  return f.v;
}
__device__ __forceinline__ v16bf gblPK(const unsigned short* p) {
  Frag f;
  f.u[0] = *(const uint4*)(p);
  f.u[1] = *(const uint4*)(p + 8);
  return f.v;
}
// A-operand with K=16..31 zeroed (for K=16 attn matmul padded to K=32)
__device__ __forceinline__ v16bf ldsA_half(const unsigned short* p) {
  Frag f;
  f.u[0] = *(const uint4*)(p);
  f.u[1] = make_uint4(0u, 0u, 0u, 0u);
  return f.v;
}
// B-operand from a 16-entry (K=t) transposed-V row; g picks t 0..7 or 8..15; elements 8..15 (K>=16) => 0.
// Branchless (data select -> v_cndmask), avoids exec-mask divergence.
__device__ __forceinline__ v16bf ldsB_vrow(const unsigned short* p, int g) {
  Frag f;
  uint4 a0 = *(const uint4*)(p);
  uint4 a1 = *(const uint4*)(p + 8);
  uint4 z  = make_uint4(0u, 0u, 0u, 0u);
  f.u[0] = (g == 0) ? a0 : a1;
  f.u[1] = z;
  return f.v;
}

__device__ __forceinline__ v8f wmma_bf16(v16bf a, v16bf b, v8f c) {
  // (neg_a, A, neg_b, B, c_mod, C, reuse_a, reuse_b)
  return __builtin_amdgcn_wmma_f32_16x16x32_bf16(false, a, false, b, (short)0, c,
                                                 false, false);
}

// ---------------------------------------------------------------------------
// Kernel 1: pre-swizzle weights (fp32 row-major [N][K=256]) into bf16
// B-operand fragment layout: pk[(nt*8 + kstep)*512 + lane*16 + e],
// value = W[(nt*16 + lane%16)*256 + kstep*32 + (e/8)*16 + (lane/16)*8 + e%8].
// Region 0: w_qkv (48 N-tiles), region 1 (offset 196608 shorts): w_out (16).
// ---------------------------------------------------------------------------
__global__ void __launch_bounds__(256) pack_weights_kernel(
    const float* __restrict__ wqkv, const float* __restrict__ wout,
    unsigned short* __restrict__ pk)
{
  int idx = blockIdx.x * 256 + threadIdx.x;     // 262144 total
  bool is_out = idx >= 196608;
  int local = is_out ? idx - 196608 : idx;
  int e    = local & 15;
  int lane = (local >> 4) & 31;
  int kt   = (local >> 9) & 7;
  int nt   = local >> 12;
  int K = kt * 32 + (e >> 3) * 16 + (lane >> 4) * 8 + (e & 7);
  int N = nt * 16 + (lane & 15);
  const float* W = is_out ? wout : wqkv;
  pk[idx] = f2bf(W[N * 256 + K]);
}

// ---------------------------------------------------------------------------
// Kernel 2: fused QKV-proj -> attention -> out-proj.
// One workgroup (8 waves) per (b,h,w); each wave owns one head in attention.
// ---------------------------------------------------------------------------
// LDS layout (ushort units):
//   [0,     4096)  Xbf   [16 t][256 c]    bf16
//   [4096,  8192)  Qbf   [16 t][256]      bf16 (pre-scaled by dh^-0.5)
//   [8192, 12288)  Kbf   [16 t][256]      bf16
//   [12288,16384)  Vt    [256 d][16 t]    bf16 (transposed)
//   [16384,20480)  Obf   [16 t][256 e]    bf16
//   [20480,24576)  attn  [8 head][16][32] bf16
//   [24576,32768)  xstage [16][256]       fp32 (async-load staging)
__global__ void __launch_bounds__(256) attn_fused_kernel(
    const float* __restrict__ x, const float* __restrict__ rel,
    const float* __restrict__ bqkv, const float* __restrict__ bout,
    const unsigned short* __restrict__ wpk, float* __restrict__ out)
{
  __shared__ __align__(16) unsigned short lds[32768];   // 64 KB
  const int tid  = threadIdx.x;
  const int lane = tid & 31;
  const int wave = tid >> 5;
  const int g    = lane >> 4;      // lane half
  const int nl   = lane & 15;
  const int wg = blockIdx.x;       // 4096 = B*H*W
  const int w  = wg & 31, h = (wg >> 5) & 31, b = wg >> 10;
  const int hw = h * 32 + w;

  // ---- Phase 0: async gather x[b,:,:,h,w] (stride H*W in c) into LDS ------
  float* xstage = (float*)&lds[24576];
  for (int i = 0; i < 16; ++i) {
    int e = tid + (i << 8);
    int t = e >> 8, c = e & 255;
    const float* gp = x + ((size_t)((b * 16 + t) * 256 + c)) * 1024 + hw;
    uint64_t ga = (uint64_t)(uintptr_t)gp;
    unsigned int la = (unsigned int)(uintptr_t)(xstage + e);
    asm volatile("global_load_async_to_lds_b32 %0, %1, off"
                 :: "v"(la), "v"(ga) : "memory");
  }
  asm volatile("s_wait_asynccnt 0x0" ::: "memory");
  __syncthreads();
  for (int i = 0; i < 16; ++i) {          // fp32 -> bf16 A-matrix
    int e = tid + (i << 8);
    lds[e] = f2bf(xstage[e]);
  }
  __syncthreads();

  // ---- Phase 1: QKV projection: [16 x 256] @ [256 x 768] ------------------
  const float qscale = 0.17677669529663687f;   // 32^-0.5
  for (int it = 0; it < 6; ++it) {
    int nt = wave + it * 8;                     // o-tile 0..47
    float bias = bqkv[nt * 16 + nl];
    v8f acc;
    for (int r = 0; r < 8; ++r) acc[r] = bias;  // bias depends only on N
    for (int ks = 0; ks < 8; ++ks) {
      v16bf a  = ldsA(&lds[nl * 256 + ks * 32 + g * 8]);
      v16bf bm = gblPK(wpk + (size_t)(nt * 8 + ks) * 512 + lane * 16);
      acc = wmma_bf16(a, bm, acc);
    }
    // Region select is wave-uniform (16-wide tile never straddles Q/K/V):
    // force a scalar branch so stores below are branch-free.
    int snt = __builtin_amdgcn_readfirstlane(nt);
    if (snt < 16) {                    // ---- Q (scaled), rows t stride 256
      unsigned short* dq = &lds[4096 + nt * 16 + nl + (8 * g) * 256];
      for (int r = 0; r < 8; ++r) dq[r * 256] = f2bf(acc[r] * qscale);
    } else if (snt < 32) {             // ---- K, rows t stride 256
      unsigned short* dk = &lds[8192 + (nt - 16) * 16 + nl + (8 * g) * 256];
      for (int r = 0; r < 8; ++r) dk[r * 256] = f2bf(acc[r]);
    } else {                           // ---- V^T: t-contiguous, pack pairs
      unsigned short* dv = &lds[12288 + ((nt - 32) * 16 + nl) * 16 + 8 * g];
      for (int r = 0; r < 8; r += 2) {
        unsigned p2 = (unsigned)f2bf(acc[r]) | ((unsigned)f2bf(acc[r + 1]) << 16);
        *(unsigned*)(dv + r) = p2;     // ds_store_b32
      }
    }
  }
  __syncthreads();

  // ---- Phase 2: per-head attention (wave == head) -------------------------
  {
    const int head = wave;
    v8f sim;                                    // C := rel_pos fragment
    for (int r = 0; r < 8; ++r)
      sim[r] = rel[head * 256 + (r + 8 * g) * 16 + nl];
    v16bf aq = ldsA(&lds[4096 + nl * 256 + head * 32 + g * 8]);
    v16bf bk = ldsB(&lds[8192 + nl * 256 + head * 32 + g * 8]);
    sim = wmma_bf16(aq, bk, sim);               // one 16x16x32 WMMA per head

    // softmax over j (= N = 16 lanes of this half)
    for (int r = 0; r < 8; ++r) {
      float m = sim[r];
      for (int d = 1; d < 16; d <<= 1) m = fmaxf(m, __shfl_xor(m, d, 32));
      float ev = __expf(sim[r] - m);
      float s = ev;
      for (int d = 1; d < 16; d <<= 1) s += __shfl_xor(s, d, 32);
      sim[r] = ev / s;
    }
    unsigned short* at = &lds[20480 + head * 512];
    for (int r = 0; r < 8; ++r)
      at[(r + 8 * g) * 32 + nl] = f2bf(sim[r]);

    // out = attn @ V  (K=16 padded to 32 with zeroed register halves)
    for (int ntile = 0; ntile < 2; ++ntile) {
      v16bf aa = ldsA_half(at + nl * 32 + g * 8);
      v16bf bv = ldsB_vrow(&lds[12288 + (head * 32 + ntile * 16 + nl) * 16], g);
      v8f o8;
      for (int r = 0; r < 8; ++r) o8[r] = 0.0f;
      o8 = wmma_bf16(aa, bv, o8);
      for (int r = 0; r < 8; ++r)
        lds[16384 + (r + 8 * g) * 256 + head * 32 + ntile * 16 + nl] = f2bf(o8[r]);
    }
  }
  __syncthreads();

  // ---- Phase 3: output projection [16 x 256] @ [256 x 256] + store --------
  for (int it = 0; it < 2; ++it) {
    int nt = wave + it * 8;                     // c-tile 0..15
    float bias = bout[nt * 16 + nl];
    v8f acc;
    for (int r = 0; r < 8; ++r) acc[r] = bias;
    for (int ks = 0; ks < 8; ++ks) {
      v16bf a  = ldsA(&lds[16384 + nl * 256 + ks * 32 + g * 8]);
      v16bf bm = gblPK(wpk + 196608 + (size_t)(nt * 8 + ks) * 512 + lane * 16);
      acc = wmma_bf16(a, bm, acc);
    }
    for (int r = 0; r < 8; ++r) {
      int t = r + 8 * g;
      int c = nt * 16 + nl;
      out[((size_t)((b * 16 + t) * 256 + c)) * 1024 + hw] = acc[r];
    }
  }
}

// ---------------------------------------------------------------------------
// Launcher. Inputs: x, rel_pos, w_qkv, b_qkv, w_out, b_out (all fp32).
// d_ws: 524288 bytes of packed bf16 weight fragments (written every call).
// ---------------------------------------------------------------------------
extern "C" void kernel_launch(void* const* d_in, const int* in_sizes, int n_in,
                              void* d_out, int out_size, void* d_ws, size_t ws_size,
                              hipStream_t stream) {
  const float* x    = (const float*)d_in[0];
  const float* rel  = (const float*)d_in[1];
  const float* wqkv = (const float*)d_in[2];
  const float* bqkv = (const float*)d_in[3];
  const float* wout = (const float*)d_in[4];
  const float* bout = (const float*)d_in[5];
  float* out = (float*)d_out;
  unsigned short* wpk = (unsigned short*)d_ws;

  pack_weights_kernel<<<1024, 256, 0, stream>>>(wqkv, wout, wpk);
  attn_fused_kernel<<<4096, 256, 0, stream>>>(x, rel, bqkv, bout, wpk, out);
}
